// LSTM_91087666414120
// MI455X (gfx1250) — matrix operand order repaired and checked by reading, not compiled, not run on hardware
//
#include <hip/hip_runtime.h>

// ---------------------------------------------------------------------------
// 2-layer LSTM + FC for MI455X (gfx1250), bf16 WMMA with f32 accumulate.
// Cell state lives in registers of a persistent kernel; h is exchanged via
// global memory with a SPLIT software grid barrier each timestep:
//   signal after storing h_t, wait only before consuming h_{t-1}, and the
//   (h-independent) x-projection executes inside the sync window.
// Recurrent weights (time-invariant) are staged once into LDS (ds_load_b128).
// ---------------------------------------------------------------------------

typedef __attribute__((ext_vector_type(16))) __bf16 v16bf;
typedef __attribute__((ext_vector_type(8)))  float  v8f;
typedef unsigned short bf16_t;

#define HID   1024
#define BSZ   64
#define TLEN  256
#define G4    4096
#define NWGS  64              // persistent workgroups in the recurrent kernel
#define NCOL  16              // h-columns owned per WG
#define BH    (BSZ * HID)
#define OUTD  512
#define IND   512
#define LROWS 64              // 4 gates * NCOL weight rows staged in LDS
#define LPITCH 1032           // 1024 + 8 bf16 pad -> 4-bank shift per row

// ----- helpers --------------------------------------------------------------

__device__ __forceinline__ unsigned short f2bf(float f) {
  unsigned u = __float_as_uint(f);
  u += 0x7FFFu + ((u >> 16) & 1u);   // round-to-nearest-even
  return (unsigned short)(u >> 16);
}

__device__ __forceinline__ float fast_sigmoid(float x) {
  x = fminf(fmaxf(x, -30.f), 30.f);
  return __builtin_amdgcn_rcpf(1.f + __expf(-x));
}

__device__ __forceinline__ float fast_tanh(float x) {
  x = fminf(fmaxf(x, -15.f), 15.f);
  float e = __expf(2.f * x);
  return (e - 1.f) * __builtin_amdgcn_rcpf(e + 1.f);
}

// Two 16-byte chunks per lane -> one WMMA operand (A 16x32 or B 32x16, bf16).
__device__ __forceinline__ v16bf load_pair(const bf16_t* p0, const bf16_t* p1) {
  union { uint4 q[2]; v16bf v; } u;
  u.q[0] = *(const uint4*)p0;
  u.q[1] = *(const uint4*)p1;
  return u.v;
}

__device__ __forceinline__ v8f wmma_bf16(v16bf a, v16bf b, v8f c) {
  return __builtin_amdgcn_wmma_f32_16x16x32_bf16(false, a, false, b,
                                                 (short)0, c, false, false);
}

// ----- split grid barrier across NWGS workgroups (epoch-based) --------------
// Arrivals for epoch e are cnt values (e-1)*NWGS .. e*NWGS-1; phases cannot
// overlap because every WG waits for epoch e-1 before signalling epoch e.

__device__ __forceinline__ void barrier_signal(int* cnt, int* flag, int epoch) {
  __threadfence();                 // release h stores (all threads)
  __syncthreads();                 // all threads of WG have stored + fenced
  if (threadIdx.x == 0) {
    int prev = __hip_atomic_fetch_add(cnt, 1, __ATOMIC_ACQ_REL,
                                      __HIP_MEMORY_SCOPE_AGENT);
    if (prev == epoch * NWGS - 1) {
      __hip_atomic_store(flag, epoch, __ATOMIC_RELEASE,
                         __HIP_MEMORY_SCOPE_AGENT);
    }
  }
}

__device__ __forceinline__ void barrier_wait(int* flag, int epoch) {
  if (threadIdx.x == 0) {
    while (__hip_atomic_load(flag, __ATOMIC_ACQUIRE,
                             __HIP_MEMORY_SCOPE_AGENT) < epoch) {
      __builtin_amdgcn_s_sleep(2);
    }
  }
  __syncthreads();
  __builtin_amdgcn_fence(__ATOMIC_ACQUIRE, "agent");   // invalidate L0/L1
}

// ----- conversion kernels ---------------------------------------------------

__global__ void cvt_f32_bf16(const float* __restrict__ in,
                             bf16_t* __restrict__ out, long n) {
  long i = (long)blockIdx.x * blockDim.x + threadIdx.x;
  long stride = (long)gridDim.x * blockDim.x;
  for (; i < n; i += stride) out[i] = f2bf(in[i]);
}

// Build Wcat[g][0..Kin+HID) = [ W_ih[g][:] | W_hh[g][:] ] in bf16 row-major.
__global__ void build_wcat(const float* __restrict__ Wih,
                           const float* __restrict__ Whh,
                           bf16_t* __restrict__ out, int Kin, long total) {
  const int Kcat = Kin + HID;
  long i = (long)blockIdx.x * blockDim.x + threadIdx.x;
  long stride = (long)gridDim.x * blockDim.x;
  for (; i < total; i += stride) {
    long g = i / Kcat;
    int  k = (int)(i - g * Kcat);
    float v = (k < Kin) ? Wih[g * Kin + k] : Whh[g * HID + (k - Kin)];
    out[i] = f2bf(v);
  }
}

__global__ void init_barriers(int* p, int n) {
  int i = blockIdx.x * blockDim.x + threadIdx.x;
  if (i < n) p[i] = 0;
}

// ----- recurrent LSTM layer (persistent, fused projection) ------------------
//
// gates[b, 4H] = bias + x_t @ W_ih^T + h_{t-1} @ W_hh^T  (one K=Kin+HID GEMM)
// 64 WGs x 4 waves; WG wg owns h columns [wg*16, wg*16+16); wave = batch-row
// tile. Each wave accumulates 4 gate tiles (i,f,g,o) sharing one A tile.
// The W_hh slice (64 rows x 1024 bf16 = 128KB) is staged in LDS once.

__global__ __launch_bounds__(128) void lstm_seq_kernel(
    const bf16_t* __restrict__ Aseq, long long sT, long long sB, int Kin,
    const bf16_t* __restrict__ Wcat, const float* __restrict__ bias,
    bf16_t* __restrict__ hbuf, int pingpong,
    int* barrier_cnt, int* barrier_flag) {

  __shared__ __align__(16) bf16_t wlds[LROWS * LPITCH];   // ~129 KB

  const int wg   = blockIdx.x;             // 0..63
  const int wid  = threadIdx.x >> 5;       // 0..3 = batch-row tile
  const int ln   = threadIdx.x & 31;
  const int lo   = ln & 15;
  const int hi   = ln >> 4;
  const int hcol0 = wg * NCOL;
  const int m     = wid * 16 + lo;         // A-matrix row (batch index)
  const int Kcat  = Kin + HID;

  // ---- stage W_hh slice into LDS: local row r = G*16 + col ----
  for (int r = wid; r < LROWS; r += 4) {
    int G   = r >> 4;
    int col = r & 15;
    const bf16_t* src = Wcat + (size_t)(G * HID + hcol0 + col) * Kcat + Kin;
    bf16_t* dst = wlds + r * LPITCH;
    #pragma unroll
    for (int c = 0; c < 4; ++c) {
      int e = (ln + 32 * c) * 8;           // 128 uint4 per 1024-element row
      *(uint4*)(dst + e) = *(const uint4*)(src + e);
    }
  }

  // Per-lane weight-row pointers (x-part, global) + bias for 4 gates.
  const bf16_t* wrow0; const bf16_t* wrow1;
  const bf16_t* wrow2; const bf16_t* wrow3;
  float bv0, bv1, bv2, bv3;
  {
    int g0 = 0 * HID + hcol0 + lo;  wrow0 = Wcat + (size_t)g0 * Kcat;  bv0 = bias[g0];
    int g1 = 1 * HID + hcol0 + lo;  wrow1 = Wcat + (size_t)g1 * Kcat;  bv1 = bias[g1];
    int g2 = 2 * HID + hcol0 + lo;  wrow2 = Wcat + (size_t)g2 * Kcat;  bv2 = bias[g2];
    int g3 = 3 * HID + hcol0 + lo;  wrow3 = Wcat + (size_t)g3 * Kcat;  bv3 = bias[g3];
  }
  // LDS row bases for this lane's column (one per gate), B-tile layout.
  const bf16_t* lrow0 = wlds + (0 * 16 + lo) * LPITCH;
  const bf16_t* lrow1 = wlds + (1 * 16 + lo) * LPITCH;
  const bf16_t* lrow2 = wlds + (2 * 16 + lo) * LPITCH;
  const bf16_t* lrow3 = wlds + (3 * 16 + lo) * LPITCH;

  const bf16_t* arow = Aseq + (size_t)m * sB;

  __syncthreads();                         // LDS weights ready

  v8f cst = {};                            // cell-state tile (persists over t)

  for (int t = 0; t < TLEN; ++t) {
    v8f ai = {}, af = {}, ag = {}, ao = {};

    // ---- x_t projection: independent of h -> overlaps the barrier window ---
    const bf16_t* axt = arow + (size_t)t * sT;
    #pragma unroll 2
    for (int kt = 0; kt < Kin; kt += 32) {
      v16bf a = load_pair(axt + kt + 16 * hi, axt + kt + 16 * hi + 8);
      v16bf b0 = load_pair(wrow0 + kt + 16 * hi, wrow0 + kt + 16 * hi + 8);
      v16bf b1 = load_pair(wrow1 + kt + 16 * hi, wrow1 + kt + 16 * hi + 8);
      v16bf b2 = load_pair(wrow2 + kt + 16 * hi, wrow2 + kt + 16 * hi + 8);
      v16bf b3 = load_pair(wrow3 + kt + 16 * hi, wrow3 + kt + 16 * hi + 8);
      ai = wmma_bf16(a, b0, ai);
      af = wmma_bf16(a, b1, af);
      ag = wmma_bf16(a, b2, ag);
      ao = wmma_bf16(a, b3, ao);
    }

    // ---- h_{t-1} projection: wait for h, weights from LDS (h_0==0: skip) ---
    if (t > 0) {
      barrier_wait(barrier_flag, t);       // h_{t-1} visible device-wide
      const bf16_t* hprev = pingpong ? (hbuf + (size_t)(t & 1) * BH)
                                     : (hbuf + (size_t)(t - 1) * BH);
      const bf16_t* hrow = hprev + (size_t)m * HID;
      #pragma unroll 2
      for (int kt = 0; kt < HID; kt += 32) {
        v16bf a = load_pair(hrow + kt + 16 * hi, hrow + kt + 16 * hi + 8);
        int ko = kt + 16 * hi;
        v16bf b0 = load_pair(lrow0 + ko, lrow0 + ko + 8);
        v16bf b1 = load_pair(lrow1 + ko, lrow1 + ko + 8);
        v16bf b2 = load_pair(lrow2 + ko, lrow2 + ko + 8);
        v16bf b3 = load_pair(lrow3 + ko, lrow3 + ko + 8);
        ai = wmma_bf16(a, b0, ai);
        af = wmma_bf16(a, b1, af);
        ag = wmma_bf16(a, b2, ag);
        ao = wmma_bf16(a, b3, ao);
      }
    }

    // ---- gate nonlinearities + state update (on accumulator layout) ----
    bf16_t* hout = pingpong ? (hbuf + (size_t)((t + 1) & 1) * BH)
                            : (hbuf + (size_t)t * BH);
    #pragma unroll
    for (int r = 0; r < 8; ++r) {
      float iv = fast_sigmoid(ai[r] + bv0);
      float fv = fast_sigmoid(af[r] + bv1);
      float gv = fast_tanh   (ag[r] + bv2);
      float ov = fast_sigmoid(ao[r] + bv3);
      float cv = fv * cst[r] + iv * gv;
      cst[r] = cv;
      float hv = ov * fast_tanh(cv);
      int mrow = wid * 16 + r + 8 * hi;            // D layout: M = r + 8*hi
      hout[(size_t)mrow * HID + hcol0 + lo] = f2bf(hv);
    }

    barrier_signal(barrier_cnt, barrier_flag, t + 1);   // publish h_t, no spin
  }
}

// ----- final FC: out[64,512] = h_last @ W_fc^T + b_fc -----------------------

__global__ __launch_bounds__(256) void fc_kernel(
    const bf16_t* __restrict__ h, const bf16_t* __restrict__ Wfc,
    const float* __restrict__ bfc, float* __restrict__ out) {
  const int tile = blockIdx.x * 8 + (threadIdx.x >> 5);   // 0..127
  const int ln = threadIdx.x & 31, lo = ln & 15, hi = ln >> 4;
  const int mtile = tile & 3;          // 4 row tiles of batch
  const int ntile = tile >> 2;         // 32 col tiles of OUTD
  const int m = mtile * 16 + lo;
  const int n = ntile * 16 + lo;

  const bf16_t* arow = h   + (size_t)m * HID;
  const bf16_t* wrow = Wfc + (size_t)n * HID;

  v8f acc = {};
  #pragma unroll 2
  for (int kt = 0; kt < HID; kt += 32) {
    v16bf a = load_pair(arow + kt + 16 * hi, arow + kt + 16 * hi + 8);
    v16bf b = load_pair(wrow + kt + 16 * hi, wrow + kt + 16 * hi + 8);
    acc = wmma_bf16(a, b, acc);
  }
  float bv = bfc[n];
  #pragma unroll
  for (int r = 0; r < 8; ++r) {
    int mrow = mtile * 16 + r + 8 * hi;
    out[(size_t)mrow * OUTD + ntile * 16 + lo] = acc[r] + bv;
  }
}

// ----- host-side orchestration ----------------------------------------------

extern "C" void kernel_launch(void* const* d_in, const int* in_sizes, int n_in,
                              void* d_out, int out_size, void* d_ws, size_t ws_size,
                              hipStream_t stream) {
  const float* x    = (const float*)d_in[0];   // [64,256,512]
  const float* Wih0 = (const float*)d_in[1];   // [4096,512]
  const float* Whh0 = (const float*)d_in[2];   // [4096,1024]
  const float* b0   = (const float*)d_in[3];   // [4096]
  const float* Wih1 = (const float*)d_in[4];   // [4096,1024]
  const float* Whh1 = (const float*)d_in[5];   // [4096,1024]
  const float* b1   = (const float*)d_in[6];   // [4096]
  const float* Wfc  = (const float*)d_in[7];   // [512,1024]
  const float* bfc  = (const float*)d_in[8];   // [512]
  float* out = (float*)d_out;                  // [64,512]

  char* w = (char*)d_ws;
  size_t o = 0;
  bf16_t* xbf  = (bf16_t*)(w + o); o += (size_t)BSZ * TLEN * IND * 2;   // 16 MB
  bf16_t* wc0  = (bf16_t*)(w + o); o += (size_t)G4 * (IND + HID) * 2;   // 12 MB
  bf16_t* wc1  = (bf16_t*)(w + o); o += (size_t)G4 * (HID + HID) * 2;   // 16 MB
  bf16_t* wfcb = (bf16_t*)(w + o); o += (size_t)OUTD * HID * 2;         //  1 MB
  bf16_t* ys0  = (bf16_t*)(w + o); o += (size_t)TLEN * BH * 2;          // 32 MB
  bf16_t* h1b  = (bf16_t*)(w + o); o += (size_t)2 * BH * 2;             // .25MB
  int*    bar  = (int*)(w + o);    o += 256;

  // 1) re-arm barriers (deterministic across graph replays)
  init_barriers<<<1, 64, 0, stream>>>(bar, 16);

  // 2) bf16 conversions / weight packing
  cvt_f32_bf16<<<2048, 256, 0, stream>>>(x, xbf, (long)BSZ * TLEN * IND);
  build_wcat<<<2048, 256, 0, stream>>>(Wih0, Whh0, wc0, IND,
                                       (long)G4 * (IND + HID));
  build_wcat<<<2048, 256, 0, stream>>>(Wih1, Whh1, wc1, HID,
                                       (long)G4 * (HID + HID));
  cvt_f32_bf16<<<512, 256, 0, stream>>>(Wfc, wfcb, (long)OUTD * HID);

  // 3) layer 0: Aseq = x [b][t][k] -> sT=IND, sB=TLEN*IND; h history = ys0[t]
  lstm_seq_kernel<<<NWGS, 128, 0, stream>>>(
      xbf, (long long)IND, (long long)TLEN * IND, IND,
      wc0, b0, ys0, /*pingpong=*/0, bar + 0, bar + 1);

  // 4) layer 1: Aseq = ys0 [t][b][k] -> sT=BH, sB=HID; ping-pong h buffer
  lstm_seq_kernel<<<NWGS, 128, 0, stream>>>(
      ys0, (long long)BH, (long long)HID, HID,
      wc1, b1, h1b, /*pingpong=*/1, bar + 2, bar + 3);

  // 5) FC on h_last (T=256 even -> final h at parity 0)
  fc_kernel<<<16, 256, 0, stream>>>(h1b, wfcb, bfc, out);
}
